// TemporalAttention_38912403701946
// MI455X (gfx1250) — matrix operand order repaired and checked by reading, not compiled, not run on hardware
//
#include <hip/hip_runtime.h>

#define B_ 8
#define L_ 8192
#define H_ 16
#define E_ 64
#define G_ 25
#define TL_ 64
#define SCALE_ 0.25f   // H^-0.5, faithful to reference (uses n_heads)
#define RECSZ_ 1664    // floats per stage-1 partial record (1600 O + 25 m + 25 s, padded)

typedef __bf16 v16bf __attribute__((ext_vector_type(16)));
typedef float  v8f   __attribute__((ext_vector_type(8)));

union ABfrag { v16bf v; unsigned int u[8]; };
union Cfrag  { v8f v; float f[8]; };

__device__ __forceinline__ unsigned short f32_bf16(float x) {
  unsigned int u = __builtin_bit_cast(unsigned int, x);
  u += 0x7FFFu + ((u >> 16) & 1u);          // round-to-nearest-even
  return (unsigned short)(u >> 16);
}
__device__ __forceinline__ unsigned int pack2_bf16(float lo, float hi) {
  return (unsigned int)f32_bf16(lo) | ((unsigned int)f32_bf16(hi) << 16);
}

// 16-bit A-matrix 16x32 (MxK) layout, ISA 7.12.2:
// lane<16: V0..3 hold K pairs 0..3, V4..7 hold pairs 8..11 ; lane>=16: +4 pairs
__device__ __forceinline__ void load_a_frag(ABfrag& a, const unsigned int* rowPtr,
                                            int kpBase, int lhi) {
#pragma unroll
  for (int j = 0; j < 8; ++j) {
    int kp = ((j < 4) ? j : j + 4) + (lhi << 2) + kpBase;
    a.u[j] = rowPtr[kp];
  }
}
// 16-bit B-matrix 32x16 (KxN), same K order as A: lane<16: V0..3 hold K pairs 0..3,
// V4..7 hold pairs 8..11 of column N=lane; lane>=16: +4 pairs, column N=lane-16
__device__ __forceinline__ void load_b_frag(ABfrag& b, const unsigned int* colPtr,
                                            int kpBase, int lhi) {
#pragma unroll
  for (int j = 0; j < 8; ++j)
    b.u[j] = colPtr[kpBase + ((j < 4) ? j : j + 4) + (lhi << 2)];
}

// ---- Stage 1: guidance attends to K/V, online softmax over an L-chunk ------
__global__ __launch_bounds__(256) void ta_stage1(
    const float* __restrict__ keys, const float* __restrict__ values,
    const float* __restrict__ gtok, float* __restrict__ gv_out,
    float* __restrict__ part, int nChunks, int chunkLen) {
  __shared__ unsigned int gK[32 * 34];       // scaled guidance bf16 [g][e/2]
  __shared__ unsigned int kb[TL_ * 34];      // K tile bf16 [l][e/2]
  __shared__ unsigned int vbT[E_ * 34];      // V^T tile bf16 [e][l/2] (pairs along l)
  __shared__ float        sb[32 * 68];       // f32 scores [g][l]
  __shared__ unsigned int pb[32 * 34];       // P bf16 [g][l/2]
  __shared__ float rowmax[32], rowsum[32], rfac[32];

  const int bh = blockIdx.x % (B_ * H_);
  const int chunk = blockIdx.x / (B_ * H_);
  const int b = bh / H_, h = bh % H_;
  const int lStart = chunk * chunkLen;
  const int lEnd = lStart + chunkLen;
  const int tid = threadIdx.x;
  const int lane = tid & 31, wave = tid >> 5;
  const int lm = lane & 15, lhi = lane >> 4;
  const int m_t = wave >> 2;                 // guidance row tile 0..1
  const int n_t = wave & 3;                  // score col / output E tile 0..3

  const size_t rowStride = (size_t)H_ * E_;
  const float* kBase = keys   + (size_t)b * L_ * rowStride + (size_t)h * E_;
  const float* vBase = values + (size_t)b * L_ * rowStride + (size_t)h * E_;

  for (int idx = tid; idx < 32 * 16; idx += 256) {
    int g = idx >> 4, c4 = idx & 15;
    float4 t = make_float4(0.f, 0.f, 0.f, 0.f);
    if (g < G_)
      t = *(const float4*)(gtok + (((size_t)b * G_ + g) * H_ + h) * E_ + c4 * 4);
    gK[g * 34 + c4 * 2]     = pack2_bf16(t.x * SCALE_, t.y * SCALE_);
    gK[g * 34 + c4 * 2 + 1] = pack2_bf16(t.z * SCALE_, t.w * SCALE_);
  }
  if (tid < 32) { rowmax[tid] = -1e30f; rowsum[tid] = 0.f; rfac[tid] = 0.f; }

  Cfrag o;
#pragma unroll
  for (int r = 0; r < 8; ++r) o.f[r] = 0.f;

  for (int lb = lStart; lb < lEnd; lb += TL_) {
    __syncthreads();                         // prior consumers done / gK ready

    if (lb + TL_ < lEnd) {                   // WGP-scope prefetch of next tile
      const size_t nrow = (size_t)(lb + TL_ + (tid >> 2)) * rowStride + (tid & 3) * 16;
      __builtin_prefetch(kBase + nrow, 0, 3);
      __builtin_prefetch(vBase + nrow, 0, 3);
    }

#pragma unroll
    for (int i = 0; i < 4; ++i) {            // K tile 64x64 f32, row-major bf16
      int idx = tid + i * 256;
      int row = idx >> 4, c4 = idx & 15;
      const float4 kv = *(const float4*)(kBase + (size_t)(lb + row) * rowStride + c4 * 4);
      kb[row * 34 + c4 * 2]     = pack2_bf16(kv.x, kv.y);
      kb[row * 34 + c4 * 2 + 1] = pack2_bf16(kv.z, kv.w);
    }
#pragma unroll
    for (int i = 0; i < 2; ++i) {            // V tile, transposed, l-pairs packed
      int idx = tid + i * 256;
      int rp = idx >> 4, c4 = idx & 15;      // rp = l/2 (0..31)
      const float* p0 = vBase + (size_t)(lb + 2 * rp) * rowStride + c4 * 4;
      const float4 v0 = *(const float4*)p0;
      const float4 v1 = *(const float4*)(p0 + rowStride);
      vbT[(c4 * 4 + 0) * 34 + rp] = pack2_bf16(v0.x, v1.x);
      vbT[(c4 * 4 + 1) * 34 + rp] = pack2_bf16(v0.y, v1.y);
      vbT[(c4 * 4 + 2) * 34 + rp] = pack2_bf16(v0.z, v1.z);
      vbT[(c4 * 4 + 3) * 34 + rp] = pack2_bf16(v0.w, v1.w);
    }
    __syncthreads();

    { // S = Gsc @ K^T : one 16x16 tile per wave, K-dim E=64 -> 2 WMMA
      Cfrag s;
#pragma unroll
      for (int r = 0; r < 8; ++r) s.f[r] = 0.f;
#pragma unroll
      for (int eb = 0; eb < 2; ++eb) {
        ABfrag a, bb;
        load_a_frag(a, gK + (m_t * 16 + lm) * 34, eb * 16, lhi);
        load_b_frag(bb, kb + (n_t * 16 + lm) * 34, eb * 16, lhi);
        s.v = __builtin_amdgcn_wmma_f32_16x16x32_bf16(false, a.v, false, bb.v,
                                                      (short)0, s.v, false, false);
      }
#pragma unroll
      for (int r = 0; r < 8; ++r)
        sb[(m_t * 16 + r + lhi * 8) * 68 + n_t * 16 + lm] = s.f[r];
    }
    __syncthreads();

    if (tid < 32) {                          // online softmax per guidance row
      const int g = tid;
      float mOld = rowmax[g];
      float tm = -1e30f;
#pragma unroll
      for (int j = 0; j < TL_; ++j) tm = fmaxf(tm, sb[g * 68 + j]);
      float mNew = fmaxf(mOld, tm);
      float fac = __expf(mOld - mNew);
      float ts = 0.f;
#pragma unroll
      for (int jp = 0; jp < TL_ / 2; ++jp) {
        float lo = __expf(sb[g * 68 + 2 * jp]     - mNew);
        float hi = __expf(sb[g * 68 + 2 * jp + 1] - mNew);
        ts += lo + hi;
        pb[g * 34 + jp] = pack2_bf16(lo, hi);
      }
      rowsum[g] = rowsum[g] * fac + ts;
      rowmax[g] = mNew;
      rfac[g] = fac;
    }
    __syncthreads();

    { // O = O*fac + P @ V : persistent accumulator tile per wave, K=64 -> 2 WMMA
#pragma unroll
      for (int r = 0; r < 8; ++r) o.f[r] *= rfac[m_t * 16 + r + lhi * 8];
#pragma unroll
      for (int kc = 0; kc < 2; ++kc) {
        ABfrag a, bb;
        load_a_frag(a, pb + (m_t * 16 + lm) * 34, kc * 16, lhi);
        load_b_frag(bb, vbT + (n_t * 16 + lm) * 34, kc * 16, lhi);
        o.v = __builtin_amdgcn_wmma_f32_16x16x32_bf16(false, a.v, false, bb.v,
                                                      (short)0, o.v, false, false);
      }
    }
  }
  __syncthreads();

  if (nChunks == 1) {                        // single pass: normalize + store gv
#pragma unroll
    for (int r = 0; r < 8; ++r) {
      int g = m_t * 16 + r + lhi * 8;
      if (g < G_)
        gv_out[((size_t)bh * G_ + g) * E_ + n_t * 16 + lm] = o.f[r] / rowsum[g];
    }
  } else {                                   // partial: unnormalized O + (m, s)
    float* rec = part + (size_t)(bh * nChunks + chunk) * RECSZ_;
#pragma unroll
    for (int r = 0; r < 8; ++r) {
      int g = m_t * 16 + r + lhi * 8;
      if (g < G_) rec[g * E_ + n_t * 16 + lm] = o.f[r];
    }
    if (tid < G_) {
      rec[G_ * E_ + tid]      = rowmax[tid];
      rec[G_ * E_ + G_ + tid] = rowsum[tid];
    }
  }
}

// ---- Stage 1b: merge per-chunk flash partials ------------------------------
__global__ __launch_bounds__(256) void ta_combine(
    const float* __restrict__ part, float* __restrict__ gv_out, int nChunks) {
  const int bh = blockIdx.x;
  for (int idx = threadIdx.x; idx < G_ * E_; idx += 256) {
    const int g = idx >> 6;
    float M = -1e30f;
    for (int c = 0; c < nChunks; ++c)
      M = fmaxf(M, part[(size_t)(bh * nChunks + c) * RECSZ_ + G_ * E_ + g]);
    float S = 0.f, acc = 0.f;
    for (int c = 0; c < nChunks; ++c) {
      const float* rec = part + (size_t)(bh * nChunks + c) * RECSZ_;
      float w = __expf(rec[G_ * E_ + g] - M);
      S   += rec[G_ * E_ + G_ + g] * w;
      acc += rec[idx] * w;
    }
    gv_out[(size_t)bh * G_ * E_ + idx] = acc / S;
  }
}

// ---- Stage 2: queries attend to guidance (exact softmax, G<=32) ------------
__global__ __launch_bounds__(256) void ta_stage2(
    const float* __restrict__ queries, const float* __restrict__ gtok,
    const float* __restrict__ gv_in, float* __restrict__ out) {
  __shared__ unsigned int gt[32 * 34];       // scaled guidance bf16 [g][e/2]
  __shared__ unsigned int gvT[E_ * 18];      // guidance_v^T bf16 [e][g/2]
  __shared__ unsigned int qb[64 * 34];       // Q tile bf16 [l][e/2]
  __shared__ float        sb2[64 * 36];      // f32 scores [l][g]
  __shared__ unsigned int pb2[64 * 18];      // P bf16 [l][g/2] (1/sum folded)

  const int nLT = L_ / 64;
  const int lt = blockIdx.x % nLT;
  const int bh = blockIdx.x / nLT;
  const int b = bh / H_, h = bh % H_;
  const int lBase = lt * 64;
  const int tid = threadIdx.x;
  const int lane = tid & 31, wave = tid >> 5;
  const int lm = lane & 15, lhi = lane >> 4;

  const size_t rowStride = (size_t)H_ * E_;
  const float* qBase = queries + (size_t)b * L_ * rowStride + (size_t)h * E_;

  for (int idx = tid; idx < 32 * 16; idx += 256) {
    int g = idx >> 4, c4 = idx & 15;
    float4 t = make_float4(0.f, 0.f, 0.f, 0.f);
    if (g < G_)
      t = *(const float4*)(gtok + (((size_t)b * G_ + g) * H_ + h) * E_ + c4 * 4);
    gt[g * 34 + c4 * 2]     = pack2_bf16(t.x * SCALE_, t.y * SCALE_);
    gt[g * 34 + c4 * 2 + 1] = pack2_bf16(t.z * SCALE_, t.w * SCALE_);
  }
  {                                          // guidance_v^T, g-pairs packed
    int gp = tid >> 4, c4 = tid & 15;        // gp 0..15 covers g 0..31
    float4 u0 = make_float4(0.f, 0.f, 0.f, 0.f), u1 = u0;
    if (2 * gp < G_)
      u0 = *(const float4*)(gv_in + ((size_t)bh * G_ + 2 * gp) * E_ + c4 * 4);
    if (2 * gp + 1 < G_)
      u1 = *(const float4*)(gv_in + ((size_t)bh * G_ + 2 * gp + 1) * E_ + c4 * 4);
    gvT[(c4 * 4 + 0) * 18 + gp] = pack2_bf16(u0.x, u1.x);
    gvT[(c4 * 4 + 1) * 18 + gp] = pack2_bf16(u0.y, u1.y);
    gvT[(c4 * 4 + 2) * 18 + gp] = pack2_bf16(u0.z, u1.z);
    gvT[(c4 * 4 + 3) * 18 + gp] = pack2_bf16(u0.w, u1.w);
  }
#pragma unroll
  for (int i = 0; i < 4; ++i) {              // Q tile 64x64 f32
    int idx = tid + i * 256;
    int row = idx >> 4, c4 = idx & 15;
    const float4 qv = *(const float4*)(qBase + (size_t)(lBase + row) * rowStride + c4 * 4);
    qb[row * 34 + c4 * 2]     = pack2_bf16(qv.x, qv.y);
    qb[row * 34 + c4 * 2 + 1] = pack2_bf16(qv.z, qv.w);
  }
  __syncthreads();

  { // S = Q @ Gsc^T : 64x32 = 8 tiles, one per wave, 2 WMMA each
    const int m_t = wave >> 1, n_t = wave & 1;
    Cfrag s;
#pragma unroll
    for (int r = 0; r < 8; ++r) s.f[r] = 0.f;
#pragma unroll
    for (int eb = 0; eb < 2; ++eb) {
      ABfrag a, bb;
      load_a_frag(a, qb + (m_t * 16 + lm) * 34, eb * 16, lhi);
      load_b_frag(bb, gt + (n_t * 16 + lm) * 34, eb * 16, lhi);
      s.v = __builtin_amdgcn_wmma_f32_16x16x32_bf16(false, a.v, false, bb.v,
                                                    (short)0, s.v, false, false);
    }
#pragma unroll
    for (int r = 0; r < 8; ++r)
      sb2[(m_t * 16 + r + lhi * 8) * 36 + n_t * 16 + lm] = s.f[r];
  }
  __syncthreads();

  if (tid < 64) {                            // exact softmax over g (mask pad)
    const int row = tid;
    float m = -1e30f;
#pragma unroll
    for (int g = 0; g < G_; ++g) m = fmaxf(m, sb2[row * 36 + g]);
    float s = 0.f;
    float p[32];
#pragma unroll
    for (int g = 0; g < 32; ++g) {
      float e = (g < G_) ? __expf(sb2[row * 36 + g] - m) : 0.f;
      p[g] = e; s += e;
    }
    float inv = 1.f / s;
#pragma unroll
    for (int kp = 0; kp < 16; ++kp)
      pb2[row * 18 + kp] = pack2_bf16(p[2 * kp] * inv, p[2 * kp + 1] * inv);
  }
  __syncthreads();

  // X = P @ Gv : 64x64 = 16 tiles, 2 per wave, single K=32 WMMA each
  float* oBase = out + (size_t)b * L_ * rowStride + (size_t)h * E_;
#pragma unroll
  for (int i = 0; i < 2; ++i) {
    int tile = wave * 2 + i;
    int m_t = tile >> 2, e_t = tile & 3;
    Cfrag oo;
#pragma unroll
    for (int r = 0; r < 8; ++r) oo.f[r] = 0.f;
    ABfrag a, bb;
    load_a_frag(a, pb2 + (m_t * 16 + lm) * 18, 0, lhi);
    load_b_frag(bb, gvT + (e_t * 16 + lm) * 18, 0, lhi);
    oo.v = __builtin_amdgcn_wmma_f32_16x16x32_bf16(false, a.v, false, bb.v,
                                                   (short)0, oo.v, false, false);
#pragma unroll
    for (int r = 0; r < 8; ++r) {
      int row = m_t * 16 + r + lhi * 8;
      oBase[(size_t)(lBase + row) * rowStride + e_t * 16 + lm] = oo.f[r];
    }
  }
}

extern "C" void kernel_launch(void* const* d_in, const int* in_sizes, int n_in,
                              void* d_out, int out_size, void* d_ws, size_t ws_size,
                              hipStream_t stream) {
  (void)in_sizes; (void)n_in; (void)out_size;
  const float* q  = (const float*)d_in[0];
  const float* k  = (const float*)d_in[1];
  const float* v  = (const float*)d_in[2];
  const float* gt = (const float*)d_in[3];
  float* out = (float*)d_out;
  float* gv  = (float*)d_ws;                          // B*H*G*E f32 = 800 KB
  float* part = gv + (size_t)B_ * H_ * G_ * E_;

  const size_t gvBytes = (size_t)B_ * H_ * G_ * E_ * sizeof(float);
  const size_t perChunkBytes = (size_t)B_ * H_ * RECSZ_ * sizeof(float);
  int C = 1;                                          // L chunks for stage 1
  if (ws_size >= gvBytes + 16 * perChunkBytes)      C = 16;
  else if (ws_size >= gvBytes + 8 * perChunkBytes)  C = 8;

  ta_stage1<<<dim3(B_ * H_ * C), dim3(256), 0, stream>>>(k, v, gt, gv, part, C, L_ / C);
  if (C > 1)
    ta_combine<<<dim3(B_ * H_), dim3(256), 0, stream>>>(part, gv, C);
  ta_stage2<<<dim3(B_ * H_ * (L_ / 64)), dim3(256), 0, stream>>>(q, gt, gv, out);
}
